// When2commFusion_27513560498208
// MI455X (gfx1250) — matrix operand order repaired and checked by reading, not compiled, not run on hardware
//
#include <hip/hip_runtime.h>
#include <hip/hip_bf16.h>
#include <math.h>

// ---------------------------------------------------------------------------
// Problem constants (from reference)
// ---------------------------------------------------------------------------
#define KB 4          // B
#define KL 5          // L
#define KC 256        // C
#define KH 64         // H
#define KW 256        // W
#define NIMG (KB*KL)  // 20 warped maps
#define NFEAT 131072  // 128 * 16 * 64

typedef __attribute__((ext_vector_type(16))) _Float16 v16h;
typedef __attribute__((ext_vector_type(8)))  float    v8f;

// CDNA5 16-bit A-fragment (16x32) K index for reg j, packed half p, lane-half lh.
// CDNA5 ISA: lanes 0-15: regs0-3 K=0..7, regs4-7 K=16..23 ; lanes 16-31: +8.
__device__ __forceinline__ int a_frag_k(int j, int p, int lh) {
    int base = (j < 4) ? (lh * 8 + 2 * j) : (16 + lh * 8 + 2 * (j - 4));
    return base + p;
}
// CDNA5 16-bit B-fragment (32x16): same K order as A fragment.
__device__ __forceinline__ int b_frag_k(int j, int p, int lh) {
    return a_frag_k(j, p, lh);
}

// ---------------------------------------------------------------------------
// Kernel 1: scatter x into (B,L) slots + bilinear affine warp (fused)
// ---------------------------------------------------------------------------
__global__ void scatter_warp(const float* __restrict__ x,
                             const int*   __restrict__ rec,
                             const float* __restrict__ ptm,   // (B,L,L,2,3)
                             float* __restrict__ warped) {
    size_t idx = (size_t)blockIdx.x * blockDim.x + threadIdx.x;
    const size_t total = (size_t)NIMG * KC * KH * KW;
    if (idx >= total) return;
    int w = (int)(idx % KW);
    int h = (int)((idx / KW) % KH);
    int c = (int)((idx / ((size_t)KW * KH)) % KC);
    int n = (int)(idx / ((size_t)KW * KH * KC));
    int b = n / KL, l = n % KL;

    int off = 0;
    #pragma unroll
    for (int i = 0; i < KB; ++i) off += (i < b) ? rec[i] : 0;
    if (l >= rec[b]) { warped[idx] = 0.0f; return; }

    const float* src = x + (((size_t)(off + l) * KC + c) * KH) * KW;
    const float* M   = ptm + (((size_t)b * KL + 0) * KL + l) * 6;  // pairwise[:,0]

    float gx = (2.0f * (float)w + 1.0f) / (float)KW - 1.0f;
    float gy = (2.0f * (float)h + 1.0f) / (float)KH - 1.0f;
    float cx = M[0] * gx + M[1] * gy + M[2];
    float cy = M[3] * gx + M[4] * gy + M[5];
    float ix = ((cx + 1.0f) * (float)KW - 1.0f) * 0.5f;
    float iy = ((cy + 1.0f) * (float)KH - 1.0f) * 0.5f;
    float x0f = floorf(ix), y0f = floorf(iy);
    float wx = ix - x0f, wy = iy - y0f;
    int x0 = (int)x0f, y0 = (int)y0f;

    auto g = [&](int yi, int xi) -> float {
        bool v = (xi >= 0) && (xi <= KW - 1) && (yi >= 0) && (yi <= KH - 1);
        int yc = yi < 0 ? 0 : (yi > KH - 1 ? KH - 1 : yi);
        int xc = xi < 0 ? 0 : (xi > KW - 1 ? KW - 1 : xi);
        float val = src[(size_t)yc * KW + xc];
        return v ? val : 0.0f;
    };
    float v00 = g(y0, x0),     v01 = g(y0, x0 + 1);
    float v10 = g(y0 + 1, x0), v11 = g(y0 + 1, x0 + 1);
    warped[idx] = (1.0f - wy) * ((1.0f - wx) * v00 + wx * v01)
                +         wy  * ((1.0f - wx) * v10 + wx * v11);
}

// ---------------------------------------------------------------------------
// Kernel 2: implicit-GEMM 3x3 conv (pad 1) + BN + ReLU, f16 WMMA, f32 accum.
// K ordered as k = r*CIN + cin (r = kh*3+kw): each 32-wide K chunk has a
// single pixel per lane -> one padding predicate per chunk, branch-free loads.
// NTILES cout-tiles share one A fragment (cuts input HBM re-reads NTILES x).
// Block = 128 thr (4 waves); wave -> 16(M) x (16*NTILES)(N) strip.
// grid = (Mtot/64, COUT/(16*NTILES), NIMG)
// ---------------------------------------------------------------------------
template<int CIN, int COUT, int STRIDE, int NTILES>
__global__ void conv3x3_bn_relu(const float* __restrict__ in,
                                const float* __restrict__ wgt,  // (COUT,CIN,3,3)
                                const float* __restrict__ bns,
                                const float* __restrict__ bnb,
                                float* __restrict__ out,
                                int Hin, int Win) {
    const int Hout = Hin / STRIDE, Wout = Win / STRIDE;
    const int Mtot = Hout * Wout;
    const size_t chs = (size_t)Hin * Win;       // channel stride (floats)

    const int lane  = threadIdx.x & 31;
    const int wave  = threadIdx.x >> 5;
    const int mBase = blockIdx.x * 64 + wave * 16;
    const int nBase = blockIdx.y * (16 * NTILES);
    const int img   = blockIdx.z;
    const int lh    = lane >> 4;
    const int row   = lane & 15;                // A: M row ; B/C: N col

    const int m  = mBase + row;
    const int ho = m / Wout, wo = m % Wout;

    const float* inN = in + (size_t)img * CIN * chs;
    // per-tile cout for this lane + weight row base
    int cout_t[NTILES];
    const float* wgtN[NTILES];
    #pragma unroll
    for (int t = 0; t < NTILES; ++t) {
        cout_t[t] = nBase + t * 16 + row;
        wgtN[t]   = wgt + (size_t)cout_t[t] * CIN * 9;  // + cin*9 + r
    }

    v8f acc[NTILES];
    #pragma unroll
    for (int t = 0; t < NTILES; ++t) acc[t] = (v8f){};

    for (int r = 0; r < 9; ++r) {
        const int kh = r / 3, kw = r - kh * 3;
        const int hi = ho * STRIDE + kh - 1;
        const int wi = wo * STRIDE + kw - 1;
        const bool valid = ((unsigned)hi < (unsigned)Hin) &&
                           ((unsigned)wi < (unsigned)Win);
        const float zmask = valid ? 1.0f : 0.0f;
        const float* aBase = inN + (size_t)(valid ? hi : 0) * Win + (valid ? wi : 0);

        for (int c0 = 0; c0 < CIN; c0 += 32) {
            v16h a;
            #pragma unroll
            for (int j = 0; j < 8; ++j) {
                #pragma unroll
                for (int p = 0; p < 2; ++p) {
                    int ko = a_frag_k(j, p, lh);                 // cin offset
                    float va = aBase[(size_t)(c0 + ko) * chs];   // unconditional
                    a[2 * j + p] = (_Float16)(zmask * va);
                }
            }
            #pragma unroll
            for (int t = 0; t < NTILES; ++t) {
                const float* wBase = wgtN[t] + r;
                v16h b;
                #pragma unroll
                for (int j = 0; j < 8; ++j) {
                    #pragma unroll
                    for (int p = 0; p < 2; ++p) {
                        int kb = b_frag_k(j, p, lh);
                        b[2 * j + p] = (_Float16)wBase[(size_t)(c0 + kb) * 9];
                    }
                }
                acc[t] = __builtin_amdgcn_wmma_f32_16x16x32_f16(
                    false, a, false, b, (short)0, acc[t], false, false);
            }
        }
    }

    // epilogue: BN + ReLU. C/D layout: lane -> N col, reg j -> M = lh*8 + j
    // (8 consecutive M rows per lane -> two b128 stores per tile)
    #pragma unroll
    for (int t = 0; t < NTILES; ++t) {
        const float s = bns[cout_t[t]], bb = bnb[cout_t[t]];
        float* dst = out + ((size_t)img * COUT + cout_t[t]) * Mtot + mBase + lh * 8;
        float4 o0, o1;
        float v0 = acc[t][0] * s + bb, v1 = acc[t][1] * s + bb;
        float v2 = acc[t][2] * s + bb, v3 = acc[t][3] * s + bb;
        float v4 = acc[t][4] * s + bb, v5 = acc[t][5] * s + bb;
        float v6 = acc[t][6] * s + bb, v7 = acc[t][7] * s + bb;
        o0.x = v0 > 0.f ? v0 : 0.f; o0.y = v1 > 0.f ? v1 : 0.f;
        o0.z = v2 > 0.f ? v2 : 0.f; o0.w = v3 > 0.f ? v3 : 0.f;
        o1.x = v4 > 0.f ? v4 : 0.f; o1.y = v5 > 0.f ? v5 : 0.f;
        o1.z = v6 > 0.f ? v6 : 0.f; o1.w = v7 > 0.f ? v7 : 0.f;
        *(float4*)(dst)     = o0;
        *(float4*)(dst + 4) = o1;
    }
}

// ---------------------------------------------------------------------------
// Kernel 3: generic GEMM  C[M,N] = act(A[M,K] @ B[K,N] + bias)
// A row-major (stride lda), B row-major (K x N). f16 WMMA, f32 accum.
// Out-of-range M rows / N cols are CLAMPED (their D rows/cols never stored).
// A: two contiguous 8-float runs per lane -> 4x b128 loads.
// B: cooperatively staged in LDS (coalesced), shared by both M-waves.
// Block = 128 thr (4 waves, 2x2 tiles of 16): grid = (ceil(N/32), ceil(M/32))
// ---------------------------------------------------------------------------
__global__ void gemm_bias_act(const float* __restrict__ A, int lda,
                              const float* __restrict__ Bm,
                              const float* __restrict__ bias,
                              float* __restrict__ C, int ldc,
                              int M, int N, int K, int relu) {
    __shared__ _Float16 Bt[32][33];   // [k][n], padded

    const int lane = threadIdx.x & 31;
    const int wave = threadIdx.x >> 5;
    const int wm = wave >> 1, wn = wave & 1;
    const int mBase = blockIdx.y * 32 + wm * 16;
    const int nb0   = blockIdx.x * 32;
    const int nBase = nb0 + wn * 16;
    const int lh  = lane >> 4;
    const int row = lane & 15;

    const int mA   = mBase + row;
    const int mAc  = mA < M ? mA : (M - 1);          // clamp
    const float* Arow = A + (size_t)mAc * lda;
    const int nCol = nBase + row;

    const int tn  = threadIdx.x & 31;
    const int tk0 = (threadIdx.x >> 5) * 8;
    const int gnc = (nb0 + tn) < N ? (nb0 + tn) : (N - 1);

    v8f acc = {};
    for (int k0 = 0; k0 < K; k0 += 32) {
        #pragma unroll
        for (int i = 0; i < 8; ++i)
            Bt[tk0 + i][tn] = (_Float16)Bm[(size_t)(k0 + tk0 + i) * N + gnc];
        __syncthreads();

        const float4 a0 = *(const float4*)(Arow + k0 + lh * 8);
        const float4 a1 = *(const float4*)(Arow + k0 + lh * 8 + 4);
        const float4 a2 = *(const float4*)(Arow + k0 + 16 + lh * 8);
        const float4 a3 = *(const float4*)(Arow + k0 + 16 + lh * 8 + 4);
        v16h a, b;
        a[0]  = (_Float16)a0.x; a[1]  = (_Float16)a0.y;
        a[2]  = (_Float16)a0.z; a[3]  = (_Float16)a0.w;
        a[4]  = (_Float16)a1.x; a[5]  = (_Float16)a1.y;
        a[6]  = (_Float16)a1.z; a[7]  = (_Float16)a1.w;
        a[8]  = (_Float16)a2.x; a[9]  = (_Float16)a2.y;
        a[10] = (_Float16)a2.z; a[11] = (_Float16)a2.w;
        a[12] = (_Float16)a3.x; a[13] = (_Float16)a3.y;
        a[14] = (_Float16)a3.z; a[15] = (_Float16)a3.w;
        #pragma unroll
        for (int j = 0; j < 8; ++j) {
            b[2 * j]     = Bt[lh * 8 + 2 * j + (j >= 4 ? 8 : 0)][wn * 16 + row];
            b[2 * j + 1] = Bt[lh * 8 + 2 * j + 1 + (j >= 4 ? 8 : 0)][wn * 16 + row];
        }
        acc = __builtin_amdgcn_wmma_f32_16x16x32_f16(
            false, a, false, b, (short)0, acc, false, false);
        __syncthreads();
    }

    const bool nvalid = (nCol < N);
    float bv = nvalid ? bias[nCol] : 0.0f;
    #pragma unroll
    for (int j = 0; j < 8; ++j) {
        int mr = mBase + lh * 8 + j;
        if (mr < M && nvalid) {
            float v = acc[j] + bv;
            if (relu) v = v > 0.0f ? v : 0.0f;
            C[(size_t)mr * ldc + nCol] = v;
        }
    }
}

// ---------------------------------------------------------------------------
// Kernel 4: scores = keys . q per (b,l), mask, exact sparsemax over L=5
// grid = B, block = 32 (one wave)
// ---------------------------------------------------------------------------
__global__ void attn_sparsemax(const float* __restrict__ keys,  // (B*L,256)
                               const float* __restrict__ qv,    // (B,256)
                               const int*   __restrict__ rec,
                               float* __restrict__ attn) {      // (B,L)
    int b = blockIdx.x;
    int lane = threadIdx.x;
    float z[KL];
    #pragma unroll
    for (int l = 0; l < KL; ++l) {
        float p = 0.0f;
        for (int k = lane; k < 256; k += 32)
            p += keys[((size_t)(b * KL + l)) * 256 + k] * qv[(size_t)b * 256 + k];
        #pragma unroll
        for (int o = 16; o > 0; o >>= 1) p += __shfl_xor(p, o, 32);
        z[l] = p;
    }
    if (lane == 0) {
        int r = rec[b];
        #pragma unroll
        for (int l = 0; l < KL; ++l) if (l >= r) z[l] = -1.0e9f;
        float zs[KL];
        #pragma unroll
        for (int l = 0; l < KL; ++l) zs[l] = z[l];
        for (int i = 1; i < KL; ++i) {           // sort descending (L=5)
            float v = zs[i]; int j = i - 1;
            while (j >= 0 && zs[j] < v) { zs[j + 1] = zs[j]; --j; }
            zs[j + 1] = v;
        }
        float csum[KL];
        float cs = 0.0f; int ksel = 0;
        #pragma unroll
        for (int l = 0; l < KL; ++l) {
            cs += zs[l]; csum[l] = cs;
            if (1.0f + (float)(l + 1) * zs[l] > cs) ksel++;
        }
        float tau = (csum[ksel - 1] - 1.0f) / (float)ksel;
        #pragma unroll
        for (int l = 0; l < KL; ++l) {
            float a = z[l] - tau;
            attn[b * KL + l] = a > 0.0f ? a : 0.0f;
        }
    }
}

// ---------------------------------------------------------------------------
// Kernel 5: out[b,:,:,:] = sum_l attn[b,l] * warped[b*L+l,:,:,:]  (float4)
// ---------------------------------------------------------------------------
__global__ void weighted_combine(const float4* __restrict__ warped,
                                 const float*  __restrict__ attn,
                                 float4* __restrict__ out) {
    size_t idx = (size_t)blockIdx.x * blockDim.x + threadIdx.x;
    const size_t per4 = (size_t)KC * KH * KW / 4;
    const size_t total = (size_t)KB * per4;
    if (idx >= total) return;
    size_t b = idx / per4, rest = idx % per4;
    float4 s; s.x = 0.f; s.y = 0.f; s.z = 0.f; s.w = 0.f;
    #pragma unroll
    for (int l = 0; l < KL; ++l) {
        float w = attn[b * KL + l];
        float4 v = warped[(b * KL + l) * per4 + rest];
        s.x += w * v.x; s.y += w * v.y; s.z += w * v.z; s.w += w * v.w;
    }
    out[idx] = s;
}

// ---------------------------------------------------------------------------
// Host orchestration
// ---------------------------------------------------------------------------
extern "C" void kernel_launch(void* const* d_in, const int* in_sizes, int n_in,
                              void* d_out, int out_size, void* d_ws, size_t ws_size,
                              hipStream_t stream) {
    (void)in_sizes; (void)n_in; (void)out_size; (void)ws_size;

    const float* x    = (const float*)d_in[0];
    const int*   rec  = (const int*)  d_in[1];
    const float* ptm  = (const float*)d_in[2];
    const float* cw[5], *cs[5], *cb[5];
    for (int i = 0; i < 5; ++i) {
        cw[i] = (const float*)d_in[3 + 3 * i];
        cs[i] = (const float*)d_in[4 + 3 * i];
        cb[i] = (const float*)d_in[5 + 3 * i];
    }
    const float* kw1 = (const float*)d_in[18]; const float* kb1 = (const float*)d_in[19];
    const float* kw2 = (const float*)d_in[20]; const float* kb2 = (const float*)d_in[21];
    const float* kw3 = (const float*)d_in[22]; const float* kb3 = (const float*)d_in[23];
    const float* qw1 = (const float*)d_in[24]; const float* qb1 = (const float*)d_in[25];
    const float* qw2 = (const float*)d_in[26]; const float* qb2 = (const float*)d_in[27];
    const float* qw3 = (const float*)d_in[28]; const float* qb3 = (const float*)d_in[29];
    const float* aw  = (const float*)d_in[30]; const float* ab  = (const float*)d_in[31];
    float* out = (float*)d_out;

    float* ws = (float*)d_ws;
    size_t off = 0;
    float* warped = ws + off; off += (size_t)NIMG * KC * KH * KW;   // 83,886,080
    float* f1 = ws + off;     off += (size_t)NIMG * 32 * 64 * 256;
    float* f2 = ws + off;     off += (size_t)NIMG * 32 * 64 * 256;
    float* f3 = ws + off;     off += (size_t)NIMG * 64 * 32 * 128;
    float* f4 = ws + off;     off += (size_t)NIMG * 64 * 32 * 128;
    float* f5 = ws + off;     off += (size_t)NIMG * 128 * 16 * 64;
    float* kh1   = ws + off;  off += (size_t)NIMG * 256;
    float* kh2   = ws + off;  off += (size_t)NIMG * 128;
    float* keysb = ws + off;  off += (size_t)NIMG * 256;
    float* qh1   = ws + off;  off += (size_t)KB * 256;
    float* qh2   = ws + off;  off += (size_t)KB * 128;
    float* qob   = ws + off;  off += (size_t)KB * 32;
    float* qvb   = ws + off;  off += (size_t)KB * 256;
    float* attnb = ws + off;  off += 32;

    // 1. scatter + warp
    {
        size_t total = (size_t)NIMG * KC * KH * KW;
        scatter_warp<<<(unsigned)((total + 255) / 256), 256, 0, stream>>>(x, rec, ptm, warped);
    }
    // 2. conv stack (implicit-GEMM WMMA); NTILES amortizes input reads
    conv3x3_bn_relu<256, 32, 1, 2><<<dim3(16384 / 64, 1, NIMG), 128, 0, stream>>>(warped, cw[0], cs[0], cb[0], f1, 64, 256);
    conv3x3_bn_relu< 32, 32, 1, 2><<<dim3(16384 / 64, 1, NIMG), 128, 0, stream>>>(f1,     cw[1], cs[1], cb[1], f2, 64, 256);
    conv3x3_bn_relu< 32, 64, 2, 4><<<dim3( 4096 / 64, 1, NIMG), 128, 0, stream>>>(f2,     cw[2], cs[2], cb[2], f3, 64, 256);
    conv3x3_bn_relu< 64, 64, 1, 4><<<dim3( 4096 / 64, 1, NIMG), 128, 0, stream>>>(f3,     cw[3], cs[3], cb[3], f4, 32, 128);
    conv3x3_bn_relu< 64,128, 2, 4><<<dim3( 1024 / 64, 2, NIMG), 128, 0, stream>>>(f4,     cw[4], cs[4], cb[4], f5, 32, 128);

    // 3. key MLP: flat = f5 viewed as (20, NFEAT)
    gemm_bias_act<<<dim3(256 / 32, 1), 128, 0, stream>>>(f5, NFEAT, kw1, kb1, kh1, 256, NIMG, 256, NFEAT, 1);
    gemm_bias_act<<<dim3(128 / 32, 1), 128, 0, stream>>>(kh1, 256, kw2, kb2, kh2, 128, NIMG, 128, 256, 1);
    gemm_bias_act<<<dim3(256 / 32, 1), 128, 0, stream>>>(kh2, 128, kw3, kb3, keysb, 256, NIMG, 256, 128, 0);

    // 4. query MLP: flat[:,0] = f5 with row stride L*NFEAT
    gemm_bias_act<<<dim3(256 / 32, 1), 128, 0, stream>>>(f5, KL * NFEAT, qw1, qb1, qh1, 256, KB, 256, NFEAT, 1);
    gemm_bias_act<<<dim3(128 / 32, 1), 128, 0, stream>>>(qh1, 256, qw2, qb2, qh2, 128, KB, 128, 256, 1);
    gemm_bias_act<<<dim3( 32 / 32, 1), 128, 0, stream>>>(qh2, 128, qw3, qb3, qob, 32, KB, 32, 128, 0);
    gemm_bias_act<<<dim3(256 / 32, 1), 128, 0, stream>>>(qob, 32, aw, ab, qvb, 256, KB, 256, 32, 0);

    // 5. attention scores + sparsemax
    attn_sparsemax<<<KB, 32, 0, stream>>>(keysb, qvb, rec, attnb);

    // 6. weighted combine
    {
        size_t total = (size_t)KB * KC * KH * KW / 4;
        weighted_combine<<<(unsigned)((total + 255) / 256), 256, 0, stream>>>(
            (const float4*)warped, attnb, (float4*)out);
    }
}
